// Retrieval_Retro_65438121722318
// MI455X (gfx1250) — compile-verified, matrix operand order checked by reading, not run on hardware
//
#include <hip/hip_runtime.h>
#include <hip/hip_bf16.h>

typedef __attribute__((ext_vector_type(16))) __bf16 v16bf;
typedef __attribute__((ext_vector_type(8)))  __bf16 v8bf;
typedef __attribute__((ext_vector_type(8)))  float  v8f;

namespace rr {

constexpr int B_    = 512;
constexpr int K_    = 3;
constexpr int G_    = 16;
constexpr int H_    = 256;
constexpr int DIN_  = 128;
constexpr int DBOND_= 128;
constexpr int OUT_  = 5000;
constexpr int LGNN_ = 3;
constexpr int LSA_  = 2;
constexpr int LCA_  = 2;
constexpr int DH_   = 32;

enum { ACT_NONE = 0, ACT_RELU = 1, ACT_PRELU = 2, ACT_SIGMOID = 3 };
// A-source modes (compile-time):
//  0: A = A1[m]                      (plain)
//  1: A = concat(A1[gidx[m]], A2[m]) (edge message: h[src] ++ eattr)
//  2: A = concat(A1[m], A2[m])       (node update: h ++ agg)
//  3: A = concat(A1[m], A2[m/div2])  (fusion: add_emb ++ main_emb broadcast)

__device__ inline __bf16 f2bf(float f) {
  union { float f; unsigned u; } v; v.f = f;
  unsigned r = v.u + 0x7FFFu + ((v.u >> 16) & 1u);   // round-to-nearest-even
  unsigned short h = (unsigned short)(r >> 16);
  return __builtin_bit_cast(__bf16, h);
}

constexpr int BM = 64, BN = 128, BK = 32;
constexpr int LDA = 40;   // padded LDS row stride (bf16 elems): 80B, 16B-aligned

// fragment load: lanes<16 take K 0-7 / 16-23, lanes>=16 take K 8-15 / 24-31
__device__ inline v16bf ldfrag(const __bf16* rowp, int base) {
  v8bf lo = *(const v8bf*)(rowp + base);
  v8bf hi = *(const v8bf*)(rowp + base + 16);
  return __builtin_shufflevector(lo, hi, 0,1,2,3,4,5,6,7,8,9,10,11,12,13,14,15);
}

__device__ inline v8bf pack8(const float* f) {
  v8bf r;
  #pragma unroll
  for (int i = 0; i < 8; ++i) r[i] = f2bf(f[i]);
  return r;
}

// CDNA5 async DMA: copy 16 bytes global -> LDS, tracked by ASYNCcnt.
// GV mode: vdst = LDS byte offset (addr[31:0] of generic shared ptr), vaddr = 64-bit VA.
__device__ inline void async_b128(const float* g, const float* l) {
  unsigned lo = (unsigned)(unsigned long long)l;
  asm volatile("global_load_async_to_lds_b128 %0, %1, off" :: "v"(lo), "v"(g) : "memory");
}
__device__ inline void wait_asynccnt0() {
  asm volatile("s_wait_asynccnt 0x0" ::: "memory");
}

template<int MODE, bool NB>
__global__ __launch_bounds__(256)
void wmma_gemm_kernel(const float* __restrict__ A1, int ka1, const int* __restrict__ gidx1,
                      const float* __restrict__ A2, int ka2, int div2,
                      const float* __restrict__ W,  const float* __restrict__ bias,
                      const float* __restrict__ residual,
                      float* __restrict__ out, const int* __restrict__ scatterIdx,
                      const float* __restrict__ alphaPtr,
                      int M, int N, int Ktot, int act)
{
  __shared__ __align__(16) __bf16 As[BM][LDA];
  __shared__ __align__(16) __bf16 Bs[BN][LDA];
  __shared__ __align__(16) float  Braw[BK][BN];   // raw f32 W tile landed by async DMA

  const int tid  = threadIdx.x;
  const int lane = tid & 31;
  const int wave = tid >> 5;
  const int wm2 = wave & 1;        // 2 row-groups of 32
  const int wn2 = wave >> 1;       // 4 col-groups of 32
  const int blockM = blockIdx.y * BM;
  const int blockN = blockIdx.x * BN;

  // ---- loop-invariant staging coordinates ----
  const int ar  = tid >> 2;          // 0..63 : A row
  const int ac0 = (tid & 3) * 8;     // 0..24 : A col chunk (8 wide)
  const int am  = blockM + ar;
  const long rowA1 = (MODE == 1) ? (long)gidx1[am] : (long)am;   // gather hoisted
  const long rowA2 = (MODE == 3) ? (long)(am / div2) : (long)am;
  const float* __restrict__ aRow1 = A1 + rowA1 * ka1;
  const float* __restrict__ aRow2 = (MODE != 0) ? (A2 + rowA2 * ka2) : nullptr;

  const int bk  = tid >> 3;          // 0..31 : B k-row
  const int bn0 = (tid & 7) * 16;    // 0..112: B col chunk (16 wide)

  v8f acc[2][2];
  #pragma unroll
  for (int a = 0; a < 2; ++a)
    #pragma unroll
    for (int b = 0; b < 2; ++b)
      #pragma unroll
      for (int i = 0; i < 8; ++i) acc[a][b][i] = 0.f;

  for (int k0 = 0; k0 < Ktot; k0 += BK) {
    __syncthreads();   // previous iteration's consumers of As/Bs/Braw are done

    if (!NB) {
      // ---- kick async DMA of the W tile (f32) into LDS; overlaps A staging ----
      const int gk = k0 + bk;
      const float* p = W + (long)gk * N + blockN + bn0;
      const float* l = &Braw[bk][bn0];
      async_b128(p,      l);
      async_b128(p + 4,  l + 4);
      async_b128(p + 8,  l + 8);
      async_b128(p + 12, l + 12);
    }

    // ---- stage A (BM x BK): chunk lies wholly in A1 or A2 (boundary % 8 == 0) ----
    {
      const int gk = k0 + ac0;
      float tmp[8];
      const float* p = (MODE == 0 || gk < ka1) ? (aRow1 + gk) : (aRow2 + (gk - ka1));
      *(float4*)&tmp[0] = *(const float4*)(p);
      *(float4*)&tmp[4] = *(const float4*)(p + 4);
      *(v8bf*)&As[ar][ac0] = pack8(tmp);
    }

    if (!NB) {
      // ---- land DMA, then convert/transpose Braw -> Bs (bf16) ----
      wait_asynccnt0();
      __syncthreads();
      float tmp[16];
      const float* p = &Braw[bk][bn0];
      *(float4*)&tmp[0]  = *(const float4*)(p);
      *(float4*)&tmp[4]  = *(const float4*)(p + 4);
      *(float4*)&tmp[8]  = *(const float4*)(p + 8);
      *(float4*)&tmp[12] = *(const float4*)(p + 12);
      #pragma unroll
      for (int j = 0; j < 16; ++j) Bs[bn0 + j][bk] = f2bf(tmp[j]);
    } else {
      // ---- guarded synchronous path for the ragged-N classifier tail ----
      const int gk = k0 + bk;
      #pragma unroll
      for (int j = 0; j < 16; ++j) {
        const int gn = blockN + bn0 + j;
        float v = (gn < N) ? W[(long)gk * N + gn] : 0.f;
        Bs[bn0 + j][bk] = f2bf(v);
      }
      if (gk + BK < Ktot)
        __builtin_prefetch(&W[(long)(gk + BK) * N + blockN + bn0], 0, 0);
    }
    __syncthreads();

    // ---- 4 WMMAs per wave per K-step (2x2 sub-tiles, fragment reuse) ----
    const int base = (lane < 16) ? 0 : 8;
    const int r16  = lane & 15;
    v16bf afrag[2], bfrag[2];
    afrag[0] = ldfrag(&As[wm2 * 32 +      r16][0], base);
    afrag[1] = ldfrag(&As[wm2 * 32 + 16 + r16][0], base);
    bfrag[0] = ldfrag(&Bs[wn2 * 32 +      r16][0], base);
    bfrag[1] = ldfrag(&Bs[wn2 * 32 + 16 + r16][0], base);
    #pragma unroll
    for (int a = 0; a < 2; ++a)
      #pragma unroll
      for (int b = 0; b < 2; ++b)
        acc[a][b] = __builtin_amdgcn_wmma_f32_16x16x32_bf16(
            false, afrag[a], false, bfrag[b], (short)0, acc[a][b], false, false);
  }

  // ---- epilogue (C/D layout: VGPR j -> M=j (+8 for lanes>=16), N=lane%16) ----
  const int r16  = lane & 15;
  const int mofs = (lane >= 16) ? 8 : 0;
  const float alpha = (act == ACT_PRELU && alphaPtr) ? alphaPtr[0] : 0.f;
  #pragma unroll
  for (int tm = 0; tm < 2; ++tm) {
    #pragma unroll
    for (int tn = 0; tn < 2; ++tn) {
      const int n = blockN + wn2 * 32 + tn * 16 + r16;
      if (NB && n >= N) continue;
      const float bv = bias ? bias[n] : 0.f;
      #pragma unroll
      for (int j = 0; j < 8; ++j) {
        const int m = blockM + wm2 * 32 + tm * 16 + mofs + j;
        float v = acc[tm][tn][j] + bv;
        if      (act == ACT_RELU)    v = v > 0.f ? v : 0.f;
        else if (act == ACT_PRELU)   v = v >= 0.f ? v : alpha * v;
        else if (act == ACT_SIGMOID) v = 1.f / (1.f + __expf(-v));
        if (residual) v += residual[(long)m * N + n];
        if (scatterIdx) atomicAdd(&out[(long)scatterIdx[m] * N + n], v);
        else            out[(long)m * N + n] = v;
      }
    }
  }
}

__global__ void zero_kernel(float* p, long n) {
  long i = (long)blockIdx.x * blockDim.x + threadIdx.x;
  long stride = (long)gridDim.x * blockDim.x;
  for (; i < n; i += stride) p[i] = 0.f;
}

// out[seg,:] = sum_{g<G} h[seg*G+g,:] * w[seg*G+g]   (batch ids contiguous blocks of G)
__global__ void pool_kernel(const float* __restrict__ h, const float* __restrict__ w,
                            float* __restrict__ out, int nSeg) {
  int i = blockIdx.x * blockDim.x + threadIdx.x;
  if (i >= nSeg * H_) return;
  int seg = i / H_, c = i % H_;
  float s = 0.f;
  #pragma unroll
  for (int g = 0; g < G_; ++g) {
    int node = seg * G_ + g;
    s += h[(long)node * H_ + c] * w[node];
  }
  out[i] = s;
}

__global__ __launch_bounds__(256)
void layernorm_kernel(const float* __restrict__ x, const float* __restrict__ g,
                      const float* __restrict__ b, float* __restrict__ y, int M) {
  __shared__ float red[256];
  const int m = blockIdx.x, t = threadIdx.x;
  if (m >= M) return;
  float v = x[(long)m * H_ + t];
  red[t] = v; __syncthreads();
  for (int s = 128; s > 0; s >>= 1) { if (t < s) red[t] += red[t + s]; __syncthreads(); }
  float mu = red[0] / H_; __syncthreads();
  float d = v - mu;
  red[t] = d * d; __syncthreads();
  for (int s = 128; s > 0; s >>= 1) { if (t < s) red[t] += red[t + s]; __syncthreads(); }
  float var = red[0] / H_;
  y[(long)m * H_ + t] = d * rsqrtf(var + 1e-5f) * g[t] + b[t];
}

// One block per sample; one wave32 per head (dh == 32 exactly). Tq,Tk <= 3.
__global__ __launch_bounds__(256)
void attention_kernel(const float* __restrict__ Q, const float* __restrict__ Km,
                      const float* __restrict__ V, float* __restrict__ O,
                      int Tq, int Tk) {
  const int b = blockIdx.x;
  const int wave = threadIdx.x >> 5;
  const int lane = threadIdx.x & 31;
  const int col = wave * DH_ + lane;
  const float scale = 0.17677669529663687f;   // 1/sqrt(32)
  for (int q = 0; q < Tq; ++q) {
    float qv = Q[((long)b * Tq + q) * H_ + col];
    float s[4];
    float mx = -1e30f;
    for (int k = 0; k < Tk; ++k) {
      float kv = Km[((long)b * Tk + k) * H_ + col];
      float p = qv * kv;
      for (int off = 16; off > 0; off >>= 1) p += __shfl_xor(p, off, 32);
      s[k] = p * scale;
      mx = fmaxf(mx, s[k]);
    }
    float den = 0.f;
    for (int k = 0; k < Tk; ++k) { s[k] = __expf(s[k] - mx); den += s[k]; }
    float o = 0.f;
    for (int k = 0; k < Tk; ++k) o += (s[k] / den) * V[((long)b * Tk + k) * H_ + col];
    O[((long)b * Tq + q) * H_ + col] = o;
  }
}

__global__ void copy_cols_kernel(const float* __restrict__ src, float* __restrict__ dst,
                                 int M, int Wsrc, int Wdst, int colOff) {
  int i = blockIdx.x * blockDim.x + threadIdx.x;
  if (i >= M * Wsrc) return;
  int m = i / Wsrc, c = i % Wsrc;
  dst[(long)m * Wdst + colOff + c] = src[(long)m * Wsrc + c];
}

// ---------------- host-side orchestration ----------------

struct Ws {
  float *xbuf, *nbuf, *qbuf, *kbuf, *vbuf, *obuf, *fbuf;
};

struct EncP {
  const float *Wf1, *Wf2, *Wk, *Wo, *Wq, *Wv;
  const float *bf1, *bf2, *bk, *bo, *bq, *bv;
  const float *ln1b, *ln1g, *ln2b, *ln2g;
};

static EncP enc_params(void* const* d_in, int base) {
  EncP e;
  e.Wf1 = (const float*)d_in[base+0];  e.Wf2 = (const float*)d_in[base+1];
  e.Wk  = (const float*)d_in[base+2];  e.Wo  = (const float*)d_in[base+3];
  e.Wq  = (const float*)d_in[base+4];  e.Wv  = (const float*)d_in[base+5];
  e.bf1 = (const float*)d_in[base+6];  e.bf2 = (const float*)d_in[base+7];
  e.bk  = (const float*)d_in[base+8];  e.bo  = (const float*)d_in[base+9];
  e.bq  = (const float*)d_in[base+10]; e.bv  = (const float*)d_in[base+11];
  e.ln1b= (const float*)d_in[base+12]; e.ln1g= (const float*)d_in[base+13];
  e.ln2b= (const float*)d_in[base+14]; e.ln2g= (const float*)d_in[base+15];
  return e;
}

static void gemm(hipStream_t s, int mode,
                 const float* A1, int ka1, const int* gidx,
                 const float* A2, int ka2, int div2,
                 const float* W, const float* bias, const float* res,
                 float* out, const int* scat, const float* alpha,
                 int M, int N, int Ktot, int act) {
  dim3 grid((N + BN - 1) / BN, (M + BM - 1) / BM);
  const bool nb = (N % BN) != 0;
  #define LAUNCH(MD, NBv) wmma_gemm_kernel<MD, NBv><<<grid, 256, 0, s>>>( \
      A1, ka1, gidx, A2, ka2, div2, W, bias, res, out, scat, alpha, M, N, Ktot, act)
  switch (mode) {
    case 0: if (nb) LAUNCH(0, true); else LAUNCH(0, false); break;
    case 1: if (nb) LAUNCH(1, true); else LAUNCH(1, false); break;
    case 2: if (nb) LAUNCH(2, true); else LAUNCH(2, false); break;
    default:if (nb) LAUNCH(3, true); else LAUNCH(3, false); break;
  }
  #undef LAUNCH
}

// pre-LN transformer encoder; kv==nullptr -> self-attention (keys = LN(x))
static void run_encoder(hipStream_t s, float* x, int Tq, const float* kv, int Tk,
                        const EncP& p, int nl, Ws& w) {
  const int Mq = B_ * Tq;
  const int Mk = kv ? B_ * Tk : Mq;
  const int Tkk = kv ? Tk : Tq;
  for (int l = 0; l < nl; ++l) {
    layernorm_kernel<<<Mq, 256, 0, s>>>(x, p.ln1g + l*H_, p.ln1b + l*H_, w.nbuf, Mq);
    const float* keys = kv ? kv : w.nbuf;
    gemm(s, 0, w.nbuf, H_, nullptr, nullptr, 0, 0, p.Wq + (long)l*H_*H_, p.bq + l*H_,
         nullptr, w.qbuf, nullptr, nullptr, Mq, H_, H_, ACT_NONE);
    gemm(s, 0, keys, H_, nullptr, nullptr, 0, 0, p.Wk + (long)l*H_*H_, p.bk + l*H_,
         nullptr, w.kbuf, nullptr, nullptr, Mk, H_, H_, ACT_NONE);
    gemm(s, 0, keys, H_, nullptr, nullptr, 0, 0, p.Wv + (long)l*H_*H_, p.bv + l*H_,
         nullptr, w.vbuf, nullptr, nullptr, Mk, H_, H_, ACT_NONE);
    attention_kernel<<<B_, 256, 0, s>>>(w.qbuf, w.kbuf, w.vbuf, w.obuf, Tq, Tkk);
    // x = x + O @ Wo + bo   (in-place residual: each (m,n) touched by one thread only)
    gemm(s, 0, w.obuf, H_, nullptr, nullptr, 0, 0, p.Wo + (long)l*H_*H_, p.bo + l*H_,
         x, x, nullptr, nullptr, Mq, H_, H_, ACT_NONE);
    layernorm_kernel<<<Mq, 256, 0, s>>>(x, p.ln2g + l*H_, p.ln2b + l*H_, w.nbuf, Mq);
    gemm(s, 0, w.nbuf, H_, nullptr, nullptr, 0, 0, p.Wf1 + (long)l*H_*4*H_, p.bf1 + l*4*H_,
         nullptr, w.fbuf, nullptr, nullptr, Mq, 4*H_, H_, ACT_RELU);
    gemm(s, 0, w.fbuf, 4*H_, nullptr, nullptr, 0, 0, p.Wf2 + (long)l*4*H_*H_, p.bf2 + l*H_,
         x, x, nullptr, nullptr, Mq, H_, 4*H_, ACT_NONE);
  }
}

static void run_gnn(hipStream_t s, const float* x, const float* eattr, const float* fcw,
                    const int* esrc, const int* edst, int nNodes, int nEdges, int nSeg,
                    const float* Wi, const float* bi, const float* Wm, const float* bm,
                    const float* Wu, const float* bu,
                    float* hA, float* hB, float* agg, float* emb) {
  gemm(s, 0, x, DIN_, nullptr, nullptr, 0, 0, Wi, bi, nullptr, hA, nullptr, nullptr,
       nNodes, H_, DIN_, ACT_NONE);
  float* hc = hA; float* hn = hB;
  for (int l = 0; l < LGNN_; ++l) {
    zero_kernel<<<2048, 256, 0, s>>>(agg, (long)nNodes * H_);
    // m = relu(concat(h[src], eattr) @ Wm + bm); agg[dst] += m  (fused scatter)
    gemm(s, 1, hc, H_, esrc, eattr, DBOND_, 0, Wm + (long)l*(H_+DBOND_)*H_, bm + l*H_,
         nullptr, agg, edst, nullptr, nEdges, H_, H_ + DBOND_, ACT_RELU);
    // h = relu(concat(h, agg) @ Wu + bu) + h
    gemm(s, 2, hc, H_, nullptr, agg, H_, 0, Wu + (long)l*2*H_*H_, bu + l*H_,
         hc, hn, nullptr, nullptr, nNodes, H_, 2*H_, ACT_RELU);
    float* t = hc; hc = hn; hn = t;
  }
  int tot = nSeg * H_;
  pool_kernel<<<(tot + 255) / 256, 256, 0, s>>>(hc, fcw, emb, nSeg);
}

static void run_branch(hipStream_t s, const float* addEmb, const float* mainEmb,
                       const float* fusW, const float* fusA, const float* fusB,
                       const EncP& sa, const EncP& ca, Ws& w, float* cout) {
  // x = prelu(concat(add_emb, main_emb[b]) @ Wfus + bfus)   [B*K, H]
  gemm(s, 3, addEmb, H_, nullptr, mainEmb, H_, K_, fusW, fusB, nullptr,
       w.xbuf, nullptr, fusA, B_ * K_, H_, 2 * H_, ACT_PRELU);
  run_encoder(s, w.xbuf, K_, nullptr, K_, sa, LSA_, w);
  hipMemcpyAsync(cout, mainEmb, (size_t)B_ * H_ * sizeof(float),
                 hipMemcpyDeviceToDevice, s);
  run_encoder(s, cout, 1, w.xbuf, K_, ca, LCA_, w);
}

} // namespace rr

extern "C" void kernel_launch(void* const* d_in, const int* in_sizes, int n_in,
                              void* d_out, int out_size, void* d_ws, size_t ws_size,
                              hipStream_t stream) {
  using namespace rr;
  if (n_in < 96) return;
  auto F = [&](int i) { return (const float*)d_in[i]; };
  auto I = [&](int i) { return (const int*)d_in[i]; };

  const float* x_main   = F(0);
  const float* ea_main  = F(1);
  const float* fcw_main = F(2);
  const float* x_add    = F(3);
  const float* ea_add   = F(4);
  const float* fcw_add  = F(5);
  const float* x_add2   = F(6);
  const float* ea_add2  = F(7);
  const float* fcw_add2 = F(8);

  const int nMain = in_sizes[0] / DIN_;
  const int eMain = in_sizes[1] / DBOND_;
  const int nAdd  = in_sizes[3] / DIN_;
  const int eAdd  = in_sizes[4] / DBOND_;

  // params pytree, flattened in sorted-key order:
  // ca1:9..24  ca2:25..40  cls:41..45  fus1:46..48  fus2:49..51  gnn:52..57
  // sa1:58..73 sa2:74..89  then edge/batch arrays 90..95
  const float* Wi = F(52); const float* Wm = F(53); const float* Wu = F(54);
  const float* bi = F(55); const float* bm = F(56); const float* bu = F(57);
  const int* es_m = I(90); const int* ed_m = I(91);
  const int* es_a = I(93); const int* ed_a = I(94);

  float* ws = (float*)d_ws;
  long o = 0;
  auto alloc = [&](long n) { float* p = ws + o; o += n; return p; };
  float* hA      = alloc((long)nAdd * H_);
  float* hB      = alloc((long)nAdd * H_);
  float* agg     = alloc((long)nAdd * H_);
  float* mainEmb = alloc((long)B_ * H_);
  float* addEmb1 = alloc((long)B_ * K_ * H_);
  float* addEmb2 = alloc((long)B_ * K_ * H_);
  Ws w;
  w.xbuf = alloc((long)B_ * K_ * H_);
  w.nbuf = alloc((long)B_ * K_ * H_);
  w.qbuf = alloc((long)B_ * K_ * H_);
  w.kbuf = alloc((long)B_ * K_ * H_);
  w.vbuf = alloc((long)B_ * K_ * H_);
  w.obuf = alloc((long)B_ * K_ * H_);
  w.fbuf = alloc((long)B_ * K_ * 4 * H_);
  float* c1 = alloc((long)B_ * H_);
  float* c2 = alloc((long)B_ * H_);
  float* z  = alloc((long)B_ * 3 * H_);
  float* zh = alloc((long)B_ * 3 * H_);
  (void)ws_size; (void)out_size;

  // --- three GNN passes + weighted graph pooling ---
  run_gnn(stream, x_main, ea_main, fcw_main, es_m, ed_m, nMain, eMain, B_,
          Wi, bi, Wm, bm, Wu, bu, hA, hB, agg, mainEmb);
  run_gnn(stream, x_add, ea_add, fcw_add, es_a, ed_a, nAdd, eAdd, B_ * K_,
          Wi, bi, Wm, bm, Wu, bu, hA, hB, agg, addEmb1);
  run_gnn(stream, x_add2, ea_add2, fcw_add2, es_a, ed_a, nAdd, eAdd, B_ * K_,
          Wi, bi, Wm, bm, Wu, bu, hA, hB, agg, addEmb2);

  // --- two retrieval branches (fusion + self-attn + cross-attn) ---
  EncP sa1 = enc_params(d_in, 58), sa2 = enc_params(d_in, 74);
  EncP ca1 = enc_params(d_in, 9),  ca2 = enc_params(d_in, 25);
  run_branch(stream, addEmb1, mainEmb, F(46), F(47), F(48), sa1, ca1, w, c1);
  run_branch(stream, addEmb2, mainEmb, F(49), F(50), F(51), sa2, ca2, w, c2);

  // --- classifier: z = concat(main, c1, c2); prelu(z@W1+b1); sigmoid(·@W2+b2) ---
  int tot = B_ * H_;
  copy_cols_kernel<<<(tot + 255) / 256, 256, 0, stream>>>(mainEmb, z, B_, H_, 3 * H_, 0);
  copy_cols_kernel<<<(tot + 255) / 256, 256, 0, stream>>>(c1,      z, B_, H_, 3 * H_, H_);
  copy_cols_kernel<<<(tot + 255) / 256, 256, 0, stream>>>(c2,      z, B_, H_, 3 * H_, 2 * H_);
  gemm(stream, 0, z, 3 * H_, nullptr, nullptr, 0, 0, F(41), F(44), nullptr,
       zh, nullptr, F(43), B_, 3 * H_, 3 * H_, ACT_PRELU);
  gemm(stream, 0, zh, 3 * H_, nullptr, nullptr, 0, 0, F(42), F(45), nullptr,
       (float*)d_out, nullptr, nullptr, B_, OUT_, 3 * H_, ACT_SIGMOID);
}
